// MultiHeadAttention_30219389895159
// MI455X (gfx1250) — matrix operand order repaired and checked by reading, not compiled, not run on hardware
//
#include <hip/hip_runtime.h>
#include <hip/hip_bf16.h>

// ---------------------------------------------------------------------------
// MHA forward for MI455X (gfx1250, wave32, WMMA).
//   x[B,T,C] fp32, W_qkv[3C,C] fp32, W_proj[C,C] fp32 -> out[B,T,C] fp32
// All four einsums run on v_wmma_f32_16x16x32_bf16 (f32 accumulate).
// Workspace: Q,K [B,H,T,Dh] bf16 ; Vt [B,H,Dh,T] bf16 ; Y [B,T,C] bf16.
// ---------------------------------------------------------------------------

typedef __attribute__((ext_vector_type(16))) __bf16 v16bf;
typedef __attribute__((ext_vector_type(8)))  __bf16 v8bf;
typedef __attribute__((ext_vector_type(8)))  float  v8f;

static constexpr int kB  = 2;
static constexpr int kT  = 2048;
static constexpr int kC  = 2048;
static constexpr int kNH = 16;
static constexpr int kDH = 128;

union AFrag { v16bf v; v8bf h[2]; };

__device__ __forceinline__ v8f wmma_bf16(v16bf a, v16bf b, v8f c) {
  return __builtin_amdgcn_wmma_f32_16x16x32_bf16(
      /*neg_a=*/false, a, /*neg_b=*/false, b,
      /*c_mod=*/(short)0, c, /*reuse_a=*/false, /*reuse_b=*/false);
}

// Stage 16 contiguous fp32 -> 16 bf16 into LDS (two 16B ds stores).
__device__ __forceinline__ void stage16_f32(const float* __restrict__ src,
                                            __bf16* dst) {
  const float4* s4 = (const float4*)src;
  float4 a = s4[0], b = s4[1], c = s4[2], d = s4[3];
  union { v8bf v; __bf16 e[8]; } u0, u1;
  u0.e[0] = (__bf16)a.x; u0.e[1] = (__bf16)a.y; u0.e[2] = (__bf16)a.z; u0.e[3] = (__bf16)a.w;
  u0.e[4] = (__bf16)b.x; u0.e[5] = (__bf16)b.y; u0.e[6] = (__bf16)b.z; u0.e[7] = (__bf16)b.w;
  u1.e[0] = (__bf16)c.x; u1.e[1] = (__bf16)c.y; u1.e[2] = (__bf16)c.z; u1.e[3] = (__bf16)c.w;
  u1.e[4] = (__bf16)d.x; u1.e[5] = (__bf16)d.y; u1.e[6] = (__bf16)d.z; u1.e[7] = (__bf16)d.w;
  *(v8bf*)dst       = u0.v;
  *(v8bf*)(dst + 8) = u1.v;
}

__device__ __forceinline__ void stage16_bf16(const __bf16* __restrict__ src,
                                             __bf16* dst) {
  *(v8bf*)dst       = *(const v8bf*)src;
  *(v8bf*)(dst + 8) = *(const v8bf*)(src + 8);
}

// ---------------------------------------------------------------------------
// Generic GEMM:  Out[M,N] = A[M,K] * W[N,K]^T   (A: fp32 or bf16, W: fp32)
// MODE 0: epilogue scatters bf16 into Q / K / Vt workspace (QKV projection).
// MODE 1: epilogue writes fp32 Out directly (output projection).
// 256 threads = 8 waves; block tile 128x128; wave tile 64x32; BK = 32.
// ---------------------------------------------------------------------------
template <int MODE, typename AT>
__global__ void __launch_bounds__(256)
mha_gemm_wmma(const AT* __restrict__ Agl, const float* __restrict__ Wgl,
              float* __restrict__ Out,
              __bf16* __restrict__ Qb, __bf16* __restrict__ Kb,
              __bf16* __restrict__ Vtb, int Kdim) {
  constexpr int LS = 40;  // padded LDS row stride (elements); 80B, 16B aligned
  __shared__ __bf16 ldsA[128 * LS];
  __shared__ __bf16 ldsB[128 * LS];

  const int tid  = threadIdx.x;
  const int lane = tid & 31;
  const int wave = tid >> 5;
  const int l15  = lane & 15;
  const int lh   = lane >> 4;
  const int wm   = wave & 1;   // 0..1 -> 64-row slab
  const int wn   = wave >> 1;  // 0..3 -> 32-col slab
  const int rowBase = blockIdx.y * 128;
  const int colBase = blockIdx.x * 128;

  v8f acc[4][2];
  const v8f zf = {0, 0, 0, 0, 0, 0, 0, 0};
#pragma unroll
  for (int mt = 0; mt < 4; ++mt)
#pragma unroll
    for (int nt = 0; nt < 2; ++nt) acc[mt][nt] = zf;

  // Staging assignment: 2 threads per tile row, 16 elements each.
  const int srow  = tid >> 1;
  const int shalf = tid & 1;
  const AT*    aSrc = Agl + (size_t)(rowBase + srow) * Kdim + shalf * 16;
  const float* bSrc = Wgl + (size_t)(colBase + srow) * Kdim + shalf * 16;
  __bf16* aDst = &ldsA[srow * LS + shalf * 16];
  __bf16* bDst = &ldsB[srow * LS + shalf * 16];

  for (int k0 = 0; k0 < Kdim; k0 += 32) {
    __syncthreads();
    if constexpr (sizeof(AT) == 4) stage16_f32((const float*)(aSrc + k0), aDst);
    else                           stage16_bf16((const __bf16*)(aSrc + k0), aDst);
    stage16_f32(bSrc + k0, bDst);
    if (k0 + 64 <= Kdim) {  // pull next K-slab toward L2/L0
      __builtin_prefetch(aSrc + k0 + 32, 0, 1);
      __builtin_prefetch(bSrc + k0 + 32, 0, 1);
    }
    __syncthreads();

    AFrag af[4], bf[2];
#pragma unroll
    for (int mt = 0; mt < 4; ++mt) {  // A: lane<16 K0-15, lane>=16 K16-31
      const int r = wm * 64 + mt * 16 + l15;
      af[mt].h[0] = *(const v8bf*)&ldsA[r * LS + lh * 16];
      af[mt].h[1] = *(const v8bf*)&ldsA[r * LS + lh * 16 + 8];
    }
#pragma unroll
    for (int nt = 0; nt < 2; ++nt) {  // B: lane = column, 16 contiguous K per half
      const int n = wn * 32 + nt * 16 + l15;
      bf[nt].h[0] = *(const v8bf*)&ldsB[n * LS + lh * 16];
      bf[nt].h[1] = *(const v8bf*)&ldsB[n * LS + lh * 16 + 8];
    }
#pragma unroll
    for (int mt = 0; mt < 4; ++mt)
#pragma unroll
      for (int nt = 0; nt < 2; ++nt)
        acc[mt][nt] = wmma_bf16(af[mt].v, bf[nt].v, acc[mt][nt]);
  }

  // Epilogue. C layout: reg r, lanes 0-15 -> (M=r, N=lane); lanes 16-31 -> (M=r+8).
#pragma unroll
  for (int mt = 0; mt < 4; ++mt) {
#pragma unroll
    for (int nt = 0; nt < 2; ++nt) {
#pragma unroll
      for (int r = 0; r < 8; ++r) {
        const int grow = rowBase + wm * 64 + mt * 16 + r + 8 * lh;
        const int gcol = colBase + wn * 32 + nt * 16 + l15;
        const float val = acc[mt][nt][r];
        if constexpr (MODE == 0) {
          const int b = grow >> 11, t = grow & (kT - 1);   // kT = 2048
          const int seg = gcol >> 11;                      // 0:Q 1:K 2:V
          const int ch = gcol & (kC - 1);
          const int h = ch >> 7, d = ch & (kDH - 1);
          const __bf16 bv = (__bf16)val;
          if (seg == 0)
            Qb[(((size_t)(b * kNH + h) * kT) + t) * kDH + d] = bv;
          else if (seg == 1)
            Kb[(((size_t)(b * kNH + h) * kT) + t) * kDH + d] = bv;
          else
            Vtb[(((size_t)(b * kNH + h) * kDH) + d) * kT + t] = bv;  // transposed
        } else {
          Out[(size_t)grow * kC + gcol] = val;
        }
      }
    }
  }
}

// ---------------------------------------------------------------------------
// Flash-attention (causal, online softmax). 128 threads = 4 waves.
// Each wave owns 16 query rows; key tiles of 32; Dh = 128.
//   S(16x32) = Q(16x128) Kt^T  : 8 WMMAs ;  O += P V : 8 WMMAs per tile.
// ---------------------------------------------------------------------------
__global__ void __launch_bounds__(128)
mha_flash_attn(const __bf16* __restrict__ Q, const __bf16* __restrict__ K,
               const __bf16* __restrict__ Vt, __bf16* __restrict__ Y) {
  __shared__ __bf16 ldsP[4 * 16 * 32];  // per-wave P tile (16x32 bf16)

  const int tid  = threadIdx.x;
  const int lane = tid & 31;
  const int wave = tid >> 5;
  const int l15  = lane & 15;
  const int lh   = lane >> 4;

  const int bh = blockIdx.y;            // b*NH + h
  const int b  = bh / kNH;
  const int h  = bh % kNH;
  const int qr0 = blockIdx.x * 64 + wave * 16;

  const __bf16* Qp  = Q  + (size_t)bh * kT * kDH;
  const __bf16* Kp  = K  + (size_t)bh * kT * kDH;
  const __bf16* Vtp = Vt + (size_t)bh * kDH * kT;
  __bf16* myP = ldsP + wave * (16 * 32);

  // Resident Q fragments: 4 chunks of K-dim 32 (A layout).
  AFrag qf[4];
#pragma unroll
  for (int dc = 0; dc < 4; ++dc) {
    const size_t base = (size_t)(qr0 + l15) * kDH + dc * 32 + lh * 16;
    qf[dc].h[0] = *(const v8bf*)(Qp + base);
    qf[dc].h[1] = *(const v8bf*)(Qp + base + 8);
  }

  const v8f zf = {0, 0, 0, 0, 0, 0, 0, 0};
  v8f o[8];
#pragma unroll
  for (int dt = 0; dt < 8; ++dt) o[dt] = zf;
  float row_m[8], row_l[8];
#pragma unroll
  for (int r = 0; r < 8; ++r) { row_m[r] = -1e30f; row_l[r] = 0.0f; }

  const float sm_scale = 0.08838834764831845f;  // 1/sqrt(128)
  const int nkt = (qr0 + 16 + 31) >> 5;         // causal key-tile count

  for (int kt = 0; kt < nkt; ++kt) {
    const int k0 = kt * 32;

    // ---- S = Q K^T (two 16x16 N-tiles, chained over Dh) ----
    v8f s0 = zf, s1 = zf;
#pragma unroll
    for (int dc = 0; dc < 4; ++dc) {
      AFrag kf0, kf1;  // B frag: lane = key column, d contiguous
      kf0.v = *(const v16bf*)(Kp + (size_t)(k0 + l15) * kDH + dc * 32 + lh * 16);
      kf1.v = *(const v16bf*)(Kp + (size_t)(k0 + 16 + l15) * kDH + dc * 32 + lh * 16);
      s0 = wmma_bf16(qf[dc].v, kf0.v, s0);
      s1 = wmma_bf16(qf[dc].v, kf1.v, s1);
    }

    // ---- scale, causal mask, online softmax, stash P in LDS ----
#pragma unroll
    for (int r = 0; r < 8; ++r) {
      const int qrow = qr0 + r + 8 * lh;
      float v0 = s0[r] * sm_scale;
      float v1 = s1[r] * sm_scale;
      if (k0 + l15 > qrow)      v0 = -1e30f;
      if (k0 + 16 + l15 > qrow) v1 = -1e30f;

      float mx = fmaxf(v0, v1);
#pragma unroll
      for (int off = 1; off < 16; off <<= 1)
        mx = fmaxf(mx, __shfl_xor(mx, off, 16));
      const float mnew = fmaxf(row_m[r], mx);
      const float corr = __expf(row_m[r] - mnew);
      const float p0 = __expf(v0 - mnew);
      const float p1 = __expf(v1 - mnew);
      float sum = p0 + p1;
#pragma unroll
      for (int off = 1; off < 16; off <<= 1)
        sum += __shfl_xor(sum, off, 16);
      row_l[r] = row_l[r] * corr + sum;
      row_m[r] = mnew;
#pragma unroll
      for (int dt = 0; dt < 8; ++dt) o[dt][r] *= corr;

      myP[(r + 8 * lh) * 32 + l15]      = (__bf16)p0;
      myP[(r + 8 * lh) * 32 + 16 + l15] = (__bf16)p1;
    }

    // Drain our DS stores before re-reading P in A-fragment layout.
    asm volatile("s_wait_dscnt 0" ::: "memory");

    AFrag pf;
    pf.h[0] = *(const v8bf*)&myP[l15 * 32 + lh * 16];
    pf.h[1] = *(const v8bf*)&myP[l15 * 32 + lh * 16 + 8];

    // ---- O += P V  (V transposed in memory -> contiguous B frags) ----
#pragma unroll
    for (int dt = 0; dt < 8; ++dt) {
      AFrag vf;  // lane = d column, keys contiguous
      vf.v = *(const v16bf*)(Vtp + (size_t)(dt * 16 + l15) * kT + k0 + lh * 16);
      o[dt] = wmma_bf16(pf.v, vf.v, o[dt]);
    }
  }

  // ---- normalize and write y[B,T,C] as bf16 ----
#pragma unroll
  for (int dt = 0; dt < 8; ++dt) {
#pragma unroll
    for (int r = 0; r < 8; ++r) {
      const int qrow = qr0 + r + 8 * lh;
      const float val = o[dt][r] / row_l[r];
      Y[((size_t)(b * kT + qrow)) * kC + h * kDH + dt * 16 + l15] = (__bf16)val;
    }
  }
}

// ---------------------------------------------------------------------------
extern "C" void kernel_launch(void* const* d_in, const int* in_sizes, int n_in,
                              void* d_out, int out_size, void* d_ws,
                              size_t ws_size, hipStream_t stream) {
  (void)in_sizes; (void)n_in; (void)out_size; (void)ws_size;
  const float* x     = (const float*)d_in[0];
  const float* Wqkv  = (const float*)d_in[1];
  const float* Wproj = (const float*)d_in[2];
  float* out = (float*)d_out;

  const size_t qkvElems = (size_t)kB * kNH * kT * kDH;  // 8,388,608
  __bf16* Qb  = (__bf16*)d_ws;
  __bf16* Kb  = Qb + qkvElems;
  __bf16* Vtb = Kb + qkvElems;
  __bf16* Yb  = Vtb + qkvElems;

  // 1) QKV projection: M=B*T=4096, N=3C=6144, K=C=2048
  dim3 g1(6144 / 128, 4096 / 128);
  mha_gemm_wmma<0, float><<<g1, 256, 0, stream>>>(
      x, Wqkv, nullptr, Qb, Kb, Vtb, kC);

  // 2) Causal flash attention per (q-tile, b*h)
  dim3 g2(kT / 64, kB * kNH);
  mha_flash_attn<<<g2, 128, 0, stream>>>(Qb, Kb, Vtb, Yb);

  // 3) Output projection: M=4096, N=C=2048, K=C=2048
  dim3 g3(kC / 128, 4096 / 128);
  mha_gemm_wmma<1, __bf16><<<g3, 256, 0, stream>>>(
      Yb, Wproj, out, nullptr, nullptr, nullptr, kC);
}
